// LSTMRelationalGraphConvolution_67336497266759
// MI455X (gfx1250) — matrix operand order repaired and checked by reading, not compiled, not run on hardware
//
#include <hip/hip_runtime.h>

typedef __attribute__((ext_vector_type(16))) __bf16    v16bf;
typedef __attribute__((ext_vector_type(8)))  float     v8f;
typedef __attribute__((ext_vector_type(8)))  _Float16  v8h;

#define NUM_NODES_C 50000
#define L_C   32
#define IN_C  64
#define HID_C 64
#define OUT_C 128
#define R_C   11
#define WAVES_C 8
#define TILES_C 3125   // 50000 / 16 exactly

__device__ __forceinline__ float sig_(float x) { return 1.0f / (1.0f + __expf(-x)); }
__device__ __forceinline__ float th_(float x)  { float e = __expf(2.0f * x); return (e - 1.0f) / (e + 1.0f); }

__device__ __forceinline__ void cvt8(v16bf& d, int base, float4 a, float4 b, float s) {
  d[base + 0] = (__bf16)(a.x * s); d[base + 1] = (__bf16)(a.y * s);
  d[base + 2] = (__bf16)(a.z * s); d[base + 3] = (__bf16)(a.w * s);
  d[base + 4] = (__bf16)(b.x * s); d[base + 5] = (__bf16)(b.y * s);
  d[base + 6] = (__bf16)(b.z * s); d[base + 7] = (__bf16)(b.w * s);
}

__device__ __forceinline__ void cvt8h(v16bf& d, int base, v8h a) {
#pragma unroll
  for (int i = 0; i < 8; ++i) d[base + i] = (__bf16)(float)a[i];
}

__global__ void zero_kernel(float* __restrict__ p, int n) {
  int i = blockIdx.x * blockDim.x + threadIdx.x;
  if (i < n) p[i] = 0.0f;
}

__global__ void relu_kernel(float* __restrict__ p, int n) {
  int i = blockIdx.x * blockDim.x + threadIdx.x;
  if (i < n) p[i] = fmaxf(p[i], 0.0f);
}

// One wave = one 16-sample tile. bf16 WMMA (16x16x32) with f32 accumulate.
// 8 waves/block, single workgroup per WGP (276 KB LDS) -> 2 waves per SIMD32.
__global__ __launch_bounds__(256, 1)
void lstm_rgc_kernel(const int*   __restrict__ neighbor_ids,
                     const int*   __restrict__ lengths,
                     const int*   __restrict__ relations,
                     const int*   __restrict__ nodes,
                     const float* __restrict__ adj_w,
                     const float* __restrict__ features,
                     const float* __restrict__ W_ih,
                     const float* __restrict__ W_hh,
                     const float* __restrict__ rgcw,
                     float*       __restrict__ out)
{
  // Untransposed bf16 weights stored flat: s_w*[row*64 + k] = W[row][k]  -> B-fragment = 2x16B contiguous per lane
  __shared__ __attribute__((aligned(32))) __bf16   s_wih[IN_C][4 * HID_C];           // 32 KB
  __shared__ __attribute__((aligned(32))) __bf16   s_whh[HID_C][4 * HID_C];          // 32 KB
  __shared__ __attribute__((aligned(32))) _Float16 s_pool[WAVES_C][R_C][16][HID_C];  // 176 KB
  __shared__ __attribute__((aligned(32))) __bf16   s_hfrag[WAVES_C][32][2][16];      // 16 KB (A-frag order)
  __shared__ __attribute__((aligned(32))) float    s_w[WAVES_C][16][L_C];            // 16 KB
  __shared__ unsigned char s_rel[WAVES_C][16][L_C];                                  // 4 KB

  const int tid = threadIdx.x;

  // Stage weights (coalesced global read, untransposed bf16 store)
  for (int i = tid; i < 4 * HID_C * IN_C; i += 32 * WAVES_C) {
    int row = i >> 6, k = i & 63;
    ((__bf16*)s_wih)[row * IN_C + k] = (__bf16)W_ih[i];
    ((__bf16*)s_whh)[row * HID_C + k] = (__bf16)W_hh[i];
  }
  __syncthreads();

  const int wv   = tid >> 5;
  const int lane = tid & 31;
  const int m16  = lane & 15;
  const int half = lane >> 4;
  const int tile = blockIdx.x * WAVES_C + wv;
  if (tile >= TILES_C) return;

  const int sample = tile * 16 + m16;
  const int len    = lengths[sample];

  // Stage masked adjacency weights + relations for this tile
  for (int t = half; t < L_C; t += 2) {
    s_w[wv][m16][t]   = (t < len) ? adj_w[sample * L_C + t] : 0.0f;
    s_rel[wv][m16][t] = (unsigned char)relations[sample * L_C + t];
  }
  for (int i = lane; i < (R_C * 16 * HID_C) / 2; i += 32) ((unsigned int*)s_pool[wv])[i] = 0u;
  for (int i = lane; i < 32 * 2 * 16; i += 32) ((__bf16*)s_hfrag[wv])[i] = (__bf16)0.0f;

  float cst[4][8];
#pragma unroll
  for (int q = 0; q < 4; ++q)
#pragma unroll
    for (int v = 0; v < 8; ++v) cst[q][v] = 0.0f;

  // ---------------- LSTM over 32 steps ----------------
  for (int t = 0; t < L_C; ++t) {
    // Compiler barrier: stop LICM from hoisting the (loop-invariant) LDS weight
    // fragments across iterations -- that blew past the VGPR budget and spilled
    // to scratch. Re-reading B-fragments from LDS each step is the cheap path.
    asm volatile("" ::: "memory");

    const bool  valid = t < len;
    const int   id    = valid ? neighbor_ids[sample * L_C + t] : 0;
    const float xs    = valid ? 1.0f : 0.0f;
    const float* fbase = features + (size_t)id * IN_C;

    // A-fragments of x_t (16x32 bf16 layout: lane=row m16; K runs depend on half)
    v16bf xA[2], hA[2];
#pragma unroll
    for (int c = 0; c < 2; ++c) {
      const int k0 = 8 * half + 32 * c;
      const int k1 = k0 + 16;
      float4 a0 = *(const float4*)(fbase + k0);
      float4 a1 = *(const float4*)(fbase + k0 + 4);
      float4 b0 = *(const float4*)(fbase + k1);
      float4 b1 = *(const float4*)(fbase + k1 + 4);
      cvt8(xA[c], 0, a0, a1, xs);
      cvt8(xA[c], 8, b0, b1, xs);
      hA[c] = *(const v16bf*)&s_hfrag[wv][lane][c][0];   // h already in A-frag order
    }

    // Per-owned-row (D-layout M = v + 8*half) adjacency weight / relation
    float wvv[8]; int rvv[8];
#pragma unroll
    for (int v = 0; v < 8; ++v) {
      const int M = v + 8 * half;
      wvv[v] = s_w[wv][M][t];
      rvv[v] = (int)s_rel[wv][M][t];
    }

#pragma unroll
    for (int q = 0; q < 4; ++q) {       // hidden column tile (16 cols each)
      v8f zacc[4];
#pragma unroll
      for (int g = 0; g < 4; ++g)
#pragma unroll
        for (int e = 0; e < 8; ++e) zacc[g][e] = 0.0f;

#pragma unroll
      for (int kc = 0; kc < 2; ++kc) {
        const int krow = kc * 32 + 8 * half;  // B-frag: lane = N, elements = K (table order)
#pragma unroll
        for (int g = 0; g < 4; ++g) {     // gates i,f,g,o -> z col tile g*4+q
          const int tcol = g * 4 + q;
          v16bf bI; { const __bf16* p_ = (const __bf16*)s_wih + (16 * tcol + m16) * IN_C + krow; uint4 lo_ = *(const uint4*)p_, hi_ = *(const uint4*)(p_ + 16); __builtin_memcpy(&bI, &lo_, 16); __builtin_memcpy((char*)&bI + 16, &hi_, 16); }
          zacc[g] = __builtin_amdgcn_wmma_f32_16x16x32_bf16(
              false, xA[kc], false, bI, (short)0, zacc[g], false, false);
          v16bf bH; { const __bf16* p_ = (const __bf16*)s_whh + (16 * tcol + m16) * HID_C + krow; uint4 lo_ = *(const uint4*)p_, hi_ = *(const uint4*)(p_ + 16); __builtin_memcpy(&bH, &lo_, 16); __builtin_memcpy((char*)&bH + 16, &hi_, 16); }
          zacc[g] = __builtin_amdgcn_wmma_f32_16x16x32_bf16(
              false, hA[kc], false, bH, (short)0, zacc[g], false, false);
        }
      }

      // Gate math + pooled accumulation + h scatter (to next step's A-frag slots)
#pragma unroll
      for (int v = 0; v < 8; ++v) {
        const float ig = sig_(zacc[0][v]);
        const float fg = sig_(zacc[1][v]);
        const float gg = th_(zacc[2][v]);
        const float og = sig_(zacc[3][v]);
        const float cn = fg * cst[q][v] + ig * gg;
        cst[q][v] = cn;
        const float h = og * th_(cn);

        const int M   = v + 8 * half;     // sample row (D layout)
        const int hid = 16 * q + m16;     // hidden index (D layout N)
        _Float16* ps = &s_pool[wv][rvv[v]][M][hid];
        *ps = (_Float16)((float)*ps + wvv[v] * h);

        // (M, hid) -> consumer lane/chunk/element in the 16-bit A layout
        const int c_ = hid >> 5;
        const int kp = hid & 31;
        int hh, e;
        if (kp < 16) { hh = kp >> 3; e = kp & 7; }
        else         { const int kq = kp & 15; hh = kq >> 3; e = 8 + (kq & 7); }
        s_hfrag[wv][M + 16 * hh][c_][e] = (__bf16)h;
      }
    }
  }

  // ---------------- RGC projection: proj = sum_r pooled_r @ W_r ----------------
  v8f acc[8];
#pragma unroll
  for (int j = 0; j < 8; ++j)
#pragma unroll
    for (int e = 0; e < 8; ++e) acc[j][e] = 0.0f;

  for (int r = 0; r < R_C; ++r) {
    v16bf pA[2];
    const _Float16* pp = &s_pool[wv][r][m16][0];
#pragma unroll
    for (int c = 0; c < 2; ++c) {
      const int k0 = 8 * half + 32 * c;
      const int k1 = k0 + 16;
      v8h a0 = *(const v8h*)(pp + k0);   // 16B LDS load
      v8h b0 = *(const v8h*)(pp + k1);
      cvt8h(pA[c], 0, a0);
      cvt8h(pA[c], 8, b0);
    }
#pragma unroll
    for (int kc = 0; kc < 2; ++kc) {
      const float* wb = rgcw + ((size_t)r * HID_C + (kc * 32 + 8 * half)) * OUT_C + m16;
#pragma unroll
      for (int j = 0; j < 8; ++j) {     // 8 output column tiles (128 cols)
        v16bf B;
#pragma unroll
        for (int s = 0; s < 16; ++s) B[s] = (__bf16)wb[(s < 8 ? s : s + 8) * OUT_C + 16 * j];
        acc[j] = __builtin_amdgcn_wmma_f32_16x16x32_bf16(
            false, pA[kc], false, B, (short)0, acc[j], false, false);
      }
    }
  }

  // Scatter-add into node accumulator (relu applied by epilogue kernel)
  int nd[8];
#pragma unroll
  for (int v = 0; v < 8; ++v) nd[v] = nodes[tile * 16 + v + 8 * half];
#pragma unroll
  for (int j = 0; j < 8; ++j)
#pragma unroll
    for (int v = 0; v < 8; ++v)
      atomicAdd(out + (size_t)nd[v] * OUT_C + 16 * j + m16, acc[j][v]);
}

extern "C" void kernel_launch(void* const* d_in, const int* in_sizes, int n_in,
                              void* d_out, int out_size, void* d_ws, size_t ws_size,
                              hipStream_t stream) {
  (void)in_sizes; (void)n_in; (void)d_ws; (void)ws_size;
  const int*   neighbor_ids = (const int*)d_in[0];
  const int*   lengths      = (const int*)d_in[1];
  const int*   relations    = (const int*)d_in[2];
  const int*   nodes        = (const int*)d_in[3];
  const float* adj_w        = (const float*)d_in[4];
  const float* features     = (const float*)d_in[5];
  const float* W_ih         = (const float*)d_in[6];
  const float* W_hh         = (const float*)d_in[7];
  const float* rgcw         = (const float*)d_in[8];
  float* out = (float*)d_out;

  const int n = NUM_NODES_C * OUT_C;   // == out_size
  zero_kernel<<<(n + 255) / 256, 256, 0, stream>>>(out, n);

  const int blocks = (TILES_C + WAVES_C - 1) / WAVES_C;   // 391
  lstm_rgc_kernel<<<blocks, WAVES_C * 32, 0, stream>>>(
      neighbor_ids, lengths, relations, nodes, adj_w, features, W_ih, W_hh, rgcw, out);

  relu_kernel<<<(n + 255) / 256, 256, 0, stream>>>(out, n);
  (void)out_size;
}
